// SAPModel_51969104281786
// MI455X (gfx1250) — matrix operand order repaired and checked by reading, not compiled, not run on hardware
//
#include <hip/hip_runtime.h>
#include <hip/hip_bf16.h>
#include <stdint.h>

typedef __attribute__((ext_vector_type(16))) _Float16 v16h;
typedef __attribute__((ext_vector_type(8)))  float    v8f;

#define BATCH          64
#define NROW           262144      // 32*32*256
#define LOG2N          18
#define BLK_ELEMS      4096        // elements scanned per workgroup
#define BLOCKS_PER_ROW 64          // NROW / BLK_ELEMS
#define THREADS        256
#define PER_THREAD     16          // BLK_ELEMS / THREADS

// workspace layout (indices in 4-byte units)
#define WS_ROWSUM_F    0                       // 64 f32 row totals
#define WS_BSUM_F      64                      // 64*64 f32 block sums
#define WS_BOFF_F      (64 + 64*64)            // 64*64 f32 exclusive block offsets
#define WS_KEEP_U      (64 + 2*64*64)          // 64*8192 u32 keep bitmask
#define KEEP_WORDS_PER_ROW 8192

// ---------------------------------------------------------------- kernel 1
// Per-block sums of relu(x). One workgroup = one 4096-element block.
__global__ void k1_block_sums(const float* __restrict__ x, float* __restrict__ ws)
{
    __shared__ float s[THREADS];
    const int row = blockIdx.x / BLOCKS_PER_ROW;
    const int blk = blockIdx.x % BLOCKS_PER_ROW;
    const int t   = threadIdx.x;
    const float* p = x + (size_t)row * NROW + (size_t)blk * BLK_ELEMS + (size_t)t * PER_THREAD;
    __builtin_prefetch(p, 0, 1);   // global_prefetch_b8
    float acc = 0.f;
    #pragma unroll
    for (int i = 0; i < PER_THREAD; i += 4) {
        float4 v = *(const float4*)(p + i);
        acc += fmaxf(v.x, 0.f) + fmaxf(v.y, 0.f) + fmaxf(v.z, 0.f) + fmaxf(v.w, 0.f);
    }
    s[t] = acc;
    __syncthreads();
    for (int off = 128; off > 0; off >>= 1) {
        if (t < off) s[t] += s[t + off];
        __syncthreads();
    }
    if (t == 0) ws[WS_BSUM_F + blockIdx.x] = s[0];
}

// ---------------------------------------------------------------- kernel 2a
// Scan the 64 block sums of each row -> exclusive block offsets + row total.
__global__ void k2a_scan_blocks(float* __restrict__ ws)
{
    __shared__ float s[BLOCKS_PER_ROW];
    const int row = blockIdx.x;
    const int t   = threadIdx.x;          // 64 threads
    const float v = ws[WS_BSUM_F + row * BLOCKS_PER_ROW + t];
    s[t] = v;
    __syncthreads();
    for (int off = 1; off < BLOCKS_PER_ROW; off <<= 1) {
        float add = (t >= off) ? s[t - off] : 0.f;
        __syncthreads();
        s[t] += add;
        __syncthreads();
    }
    ws[WS_BOFF_F + row * BLOCKS_PER_ROW + t] = s[t] - v;   // exclusive
    if (t == BLOCKS_PER_ROW - 1) ws[WS_ROWSUM_F + row] = s[t];
}

// ---------------------------------------------------------------- kernel 2b
// Intra-block inclusive scan of relu(x), offset by block offset -> cdf.
// The 256 per-thread totals are scanned with ONE v_wmma_f32_16x16x32_f16:
//   D(16x16,f32) = Lstrict(16x16 ones, f16-exact) x T(16x16 tile of totals)
// giving all 256 exclusive prefixes in a single tensor-pipe op (wave 0).
__global__ void k2b_block_scan(const float* __restrict__ x,
                               const float* __restrict__ ws,
                               float* __restrict__ cdf)
{
    __shared__ float totals[THREADS];
    __shared__ float excl[THREADS];
    __shared__ float coloff[16];

    const int row = blockIdx.x / BLOCKS_PER_ROW;
    const int blk = blockIdx.x % BLOCKS_PER_ROW;
    const int t   = threadIdx.x;
    const size_t base = (size_t)row * NROW + (size_t)blk * BLK_ELEMS + (size_t)t * PER_THREAD;

    float a[PER_THREAD];
    #pragma unroll
    for (int i = 0; i < PER_THREAD; i += 4) {
        float4 v = *(const float4*)(x + base + i);
        a[i]     = fmaxf(v.x, 0.f);
        a[i + 1] = fmaxf(v.y, 0.f);
        a[i + 2] = fmaxf(v.z, 0.f);
        a[i + 3] = fmaxf(v.w, 0.f);
    }
    float tot = 0.f;
    #pragma unroll
    for (int i = 0; i < PER_THREAD; ++i) tot += a[i];
    totals[t] = tot;
    __syncthreads();

    // ---- wave 0: matmul-scan of the 256 totals (EXEC all-1s in this wave)
    if (t < 32) {
        const int  lane  = t;
        const int  m     = lane & 15;
        const bool hi    = lane >= 16;
        const int  kbase = hi ? 8 : 0;

        // A (16x32 f16): strictly-lower-triangular ones in K=0..15, zero K=16..31.
        // ISA layout: lanes 0-15 hold K=0..7 (elems 0-7) & K=16..23 (elems 8-15);
        //             lanes 16-31 hold K=8..15 & K=24..31. M = lane%16.
        v16h amat;
        #pragma unroll
        for (int e = 0; e < 8; ++e) {
            const int k  = kbase + e;
            amat[e]      = (_Float16)((k < m) ? 1.0f : 0.0f);
            amat[e + 8]  = (_Float16)0.0f;
        }
        // B (32x16 f16): column n = totals[n*16 .. n*16+15] in K=0..15, zero K>=16.
        // ISA layout: lanes 0-15 hold K=0..7 (elems 0-7) & K=16..23 (elems 8-15); lanes 16-31 K=8..15 & K=24..31.
        const int n = lane & 15;
        v16h bmat;
        #pragma unroll
        for (int e = 0; e < 16; ++e)
            bmat[e] = (e >= 8) ? (_Float16)0.0f : (_Float16)totals[n * 16 + kbase + e];

        v8f d = {};
        d = __builtin_amdgcn_wmma_f32_16x16x32_f16(
                /*neg_a=*/false, amat, /*neg_b=*/false, bmat,
                /*c_mod=*/(short)0, d, /*reuse_a=*/false, /*reuse_b=*/false);

        // D layout: lane L elem r -> M = r + (L<16 ? 0 : 8), N = L%16.
        #pragma unroll
        for (int r2 = 0; r2 < 8; ++r2) {
            const int mm = r2 + (hi ? 8 : 0);
            excl[n * 16 + mm] = d[r2];       // exclusive prefix within column
        }
    }
    __syncthreads();

    if (t == 0) {   // 16-element cross-column exclusive scan (trivial)
        float run = 0.f;
        for (int n2 = 0; n2 < 16; ++n2) {
            coloff[n2] = run;
            run += excl[n2 * 16 + 15] + totals[n2 * 16 + 15];
        }
    }
    __syncthreads();

    float run = ws[WS_BOFF_F + row * BLOCKS_PER_ROW + blk] + excl[t] + coloff[t >> 4];
    float o[PER_THREAD];
    #pragma unroll
    for (int i = 0; i < PER_THREAD; ++i) { run += a[i]; o[i] = run; }
    #pragma unroll
    for (int i = 0; i < PER_THREAD; i += 4) {
        float4 v; v.x = o[i]; v.y = o[i + 1]; v.z = o[i + 2]; v.w = o[i + 3];
        *(float4*)(cdf + base + i) = v;
    }
}

// ---------------------------------------------------------------- kernel 3
// One thread per categorical sample: threefry2x32 -> u, binary search in the
// L2-resident unnormalized cdf (67 MB << 192 MB L2), atomicOr keep bit.
__device__ __forceinline__ uint32_t rotl32(uint32_t x, int r) {
    return (x << r) | (x >> (32 - r));
}

__global__ void k3_sample(const float* __restrict__ cdf,
                          const float* __restrict__ ws,
                          uint32_t* __restrict__ keep)
{
    const uint32_t s   = blockIdx.x * THREADS + threadIdx.x;  // sample id
    const uint32_t row = s >> LOG2N;

    // threefry2x32, key=(42,0), counter=(s,0)
    const uint32_t k0 = 42u, k1 = 0u;
    const uint32_t ks[3] = { k0, k1, 0x1BD11BDAu ^ k0 ^ k1 };
    uint32_t x0 = s + k0, x1 = k1;
    const int R0[4] = {13, 15, 26, 6}, R1[4] = {17, 29, 16, 24};
    #pragma unroll
    for (int i = 0; i < 5; ++i) {
        const int* rr = (i & 1) ? R1 : R0;
        #pragma unroll
        for (int j = 0; j < 4; ++j) { x0 += x1; x1 = rotl32(x1, rr[j]); x1 ^= x0; }
        x0 += ks[(i + 1) % 3];
        x1 += ks[(i + 2) % 3] + (uint32_t)(i + 1);
    }
    const float u      = (float)(x0 >> 8) * (1.0f / 16777216.0f);
    const float target = u * ws[WS_ROWSUM_F + row];

    const float* c = cdf + (size_t)row * NROW;
    int lo = 0, hi = NROW;                 // first index with cdf >= target
    while (lo < hi) {
        const int mid = (lo + hi) >> 1;
        if (c[mid] < target) lo = mid + 1; else hi = mid;
    }
    const int ind = (lo < NROW - 1) ? lo : (NROW - 1);
    atomicOr(&keep[row * KEEP_WORDS_PER_ROW + (ind >> 5)], 1u << (ind & 31));
}

// ---------------------------------------------------------------- kernel 4
// out = keep ? relu(x) / (1 - (1-p)^S) : 0,  p = relu(x)/rowsum
__global__ void k4_output(const float* __restrict__ x,
                          const float* __restrict__ ws,
                          const uint32_t* __restrict__ keep,
                          float* __restrict__ out)
{
    const size_t   e0  = ((size_t)blockIdx.x * THREADS + threadIdx.x) * 4;
    const uint32_t row = (uint32_t)(e0 >> LOG2N);
    const float inv_total = 1.0f / ws[WS_ROWSUM_F + row];
    const uint32_t local  = (uint32_t)(e0 & (NROW - 1));
    const uint32_t kw     = keep[row * KEEP_WORDS_PER_ROW + (local >> 5)];
    const uint32_t bit0   = local & 31u;

    float4 v = *(const float4*)(x + e0);
    float in4[4] = { v.x, v.y, v.z, v.w };
    float o4[4];
    #pragma unroll
    for (int i = 0; i < 4; ++i) {
        const float a    = fmaxf(in4[i], 0.f);
        const bool  kept = (kw >> (bit0 + i)) & 1u;
        const float p    = a * inv_total;
        const float scale = -expm1f(262144.0f * log1pf(-p));   // 1-(1-p)^S
        o4[i] = (kept && a > 0.f) ? (a / scale) : 0.f;
    }
    float4 ov; ov.x = o4[0]; ov.y = o4[1]; ov.z = o4[2]; ov.w = o4[3];
    *(float4*)(out + e0) = ov;
}

// ---------------------------------------------------------------- launch
extern "C" void kernel_launch(void* const* d_in, const int* in_sizes, int n_in,
                              void* d_out, int out_size, void* d_ws, size_t ws_size,
                              hipStream_t stream)
{
    (void)in_sizes; (void)n_in; (void)out_size; (void)ws_size;
    const float* x   = (const float*)d_in[0];
    float*       out = (float*)d_out;
    float*       wsf = (float*)d_ws;
    uint32_t*    keep = (uint32_t*)d_ws + WS_KEEP_U;

    // zero the keep bitmask (ws is poisoned; must be deterministic per call)
    hipMemsetAsync(keep, 0, (size_t)BATCH * KEEP_WORDS_PER_ROW * sizeof(uint32_t), stream);

    k1_block_sums <<<BATCH * BLOCKS_PER_ROW, THREADS,        0, stream>>>(x, wsf);
    k2a_scan_blocks<<<BATCH,                 BLOCKS_PER_ROW, 0, stream>>>(wsf);
    k2b_block_scan<<<BATCH * BLOCKS_PER_ROW, THREADS,        0, stream>>>(x, wsf, out); // cdf -> d_out (scratch)
    k3_sample     <<<(BATCH * NROW) / THREADS, THREADS,      0, stream>>>(out, wsf, keep);
    k4_output     <<<(BATCH * NROW) / (THREADS * 4), THREADS, 0, stream>>>(x, wsf, keep, out);
}
